// PillarResNet18_25958782337396
// MI455X (gfx1250) — matrix operand order repaired and checked by reading, not compiled, not run on hardware
//
#include <hip/hip_runtime.h>
#include <cstdint>
#include <cstddef>

typedef __attribute__((ext_vector_type(16))) __bf16          v16bf;
typedef __attribute__((ext_vector_type(8)))  float           v8f;
typedef __attribute__((ext_vector_type(8)))  unsigned short  v8us;

static __device__ __forceinline__ unsigned short f2bf_bits(float f) {
  unsigned u = __builtin_bit_cast(unsigned, f);
  u += 0x7fffu + ((u >> 16) & 1u);          // round-to-nearest-even
  return (unsigned short)(u >> 16);
}
static __device__ __forceinline__ float bf2f(unsigned short b) {
  unsigned u = ((unsigned)b) << 16;
  return __builtin_bit_cast(float, u);
}

// ---------------- mask = (mask_p < 0.1) ----------------
__global__ void make_mask_kernel(const float* __restrict__ mp, float* __restrict__ m, int n) {
  int i = blockIdx.x * blockDim.x + threadIdx.x;
  if (i < n) m[i] = (mp[i] < 0.1f) ? 1.0f : 0.0f;
}

// ---------------- 3x3 stride-2 pad-1 max-pool of mask ----------------
__global__ void down_mask_kernel(const float* __restrict__ mi, float* __restrict__ mo,
                                 int N, int Hi, int Wi) {
  int Ho = Hi >> 1, Wo = Wi >> 1;
  int i = blockIdx.x * blockDim.x + threadIdx.x;
  int total = N * Ho * Wo;
  if (i >= total) return;
  int wo = i % Wo; int t = i / Wo; int ho = t % Ho; int n = t / Ho;
  float v = 0.0f;
  for (int dh = -1; dh <= 1; ++dh)
    for (int dw = -1; dw <= 1; ++dw) {
      int h = 2 * ho + dh, w = 2 * wo + dw;
      if (h >= 0 && h < Hi && w >= 0 && w < Wi)
        v = fmaxf(v, mi[((long)n * Hi + h) * Wi + w]);
    }
  mo[i] = v;
}

// ---------------- zero the 1-element halo ring of a padded NHWC bf16 buffer ----------------
__global__ void ring_zero_kernel(__bf16* __restrict__ buf, int N, int H, int W, int C) {
  int Hp = H + 2, Wp = W + 2;
  long P = 2L * Wp + 2L * H;                 // ring positions per image
  long total = (long)N * P * C;
  long i = (long)blockIdx.x * blockDim.x + threadIdx.x;
  if (i >= total) return;
  int c = (int)(i % C); long p = i / C;
  int n = (int)(p / P); long q = p % P;
  int h, w;
  if (q < Wp)            { h = 0;      w = (int)q; }
  else if (q < 2L * Wp)  { h = Hp - 1; w = (int)(q - Wp); }
  else { long r = q - 2L * Wp; h = 1 + (int)(r >> 1); w = (r & 1) ? (Wp - 1) : 0; }
  buf[((long)(n * Hp + h) * Wp + w) * C + c] = __builtin_bit_cast(__bf16, (unsigned short)0);
}

// ---------------- x(NCHW f32) * mask -> padded NHWC bf16 ----------------
__global__ void pack_input_kernel(const float* __restrict__ x, const float* __restrict__ mask,
                                  __bf16* __restrict__ out, int N, int C, int H, int W) {
  long i = (long)blockIdx.x * blockDim.x + threadIdx.x;
  long total = (long)N * H * W * C;
  if (i >= total) return;
  int c = (int)(i % C); long t = i / C;
  int w = (int)(t % W); t /= W;
  int h = (int)(t % H); int n = (int)(t / H);
  float v = x[(((long)n * C + c) * H + h) * W + w] * mask[((long)n * H + h) * W + w];
  int Hp = H + 2, Wp = W + 2;
  out[((long)(n * Hp + h + 1) * Wp + (w + 1)) * C + c] = __builtin_bit_cast(__bf16, f2bf_bits(v));
}

// ---------------- pack OIHW f32 weights into WMMA A-fragment layout (bf16) ----------------
// Linear layout: (((cot*9 + tap)*nCic + c)*32 + lane)*16 + e   (512 bf16 per fragment block)
// A 16x32 order chosen here (B uses the same k order): lane = {M=lane&15, half=lane>>4}; element e -> j=e>>1,r=e&1;
//   K = 16*half + e
__global__ void pack_weights_kernel(const float* __restrict__ w, __bf16* __restrict__ wpk,
                                    int Cin, int Cout) {
  int nCic = Cin >> 5;
  long total = (long)(Cout >> 4) * 9 * nCic * 512;
  long i = (long)blockIdx.x * blockDim.x + threadIdx.x;
  if (i >= total) return;
  int e = (int)(i & 15);
  int lane = (int)((i >> 4) & 31);
  long r = i >> 9;
  int c = (int)(r % nCic); r /= nCic;
  int tap = (int)(r % 9);
  int cot = (int)(r / 9);
  int m = lane & 15, lh = lane >> 4;
  int j = e >> 1, rr = e & 1;
  int k = 16 * lh + 2 * j + rr;
  int co = cot * 16 + m;
  int ci = c * 32 + k;
  wpk[i] = __builtin_bit_cast(__bf16, f2bf_bits(w[((long)co * Cin + ci) * 9 + tap]));
}

// ---------------- fused implicit-GEMM 3x3 conv via v_wmma_f32_16x16x32_bf16 ----------------
// out = relu( (conv(in,w)*scale+shift) * mask + resid )
// in/resid/out_bf: halo-padded NHWC bf16 ((H+2)x(W+2)).  out_f32: NCHW f32 stage output.
// mask: unpadded [N,Ho,Wo] f32.  Zero halo ring makes all 9 taps unconditional.
__global__ __launch_bounds__(128) void conv3x3_wmma_kernel(
    const __bf16* __restrict__ in, const __bf16* __restrict__ wpk,
    const float* __restrict__ scale, const float* __restrict__ shift,
    const float* __restrict__ mask, const __bf16* __restrict__ resid,
    __bf16* __restrict__ out_bf, float* __restrict__ out_f32,
    int Cin, int Cout, int Hi, int Wi, int Ho, int Wo, int stride)
{
  const int wave = threadIdx.x >> 5;
  const int lane = threadIdx.x & 31;
  const int wo0 = blockIdx.x * 64 + wave * 16;
  if (wo0 >= Wo) return;                    // wave-uniform
  const int ho = blockIdx.y;
  const int cotN = Cout >> 4;
  const int cot = blockIdx.z % cotN;
  const int n = blockIdx.z / cotN;
  const int ln = lane & 15;                 // N index within tile
  const int lh = lane >> 4;                 // K/M half selector
  const int wo = wo0 + ln;
  const int nCic = Cin >> 5;
  const int Hip = Hi + 2, Wip = Wi + 2;

  v8f acc = (v8f)0.0f;

  const __bf16* wbase = wpk + (long)cot * 9 * nCic * 512 + lane * 16;
  // top-left tap address (kh=0,kw=0): padded row = ho*stride, padded col = wo*stride
  const __bf16* ibase = in + (((long)n * Hip + ho * stride) * Wip + wo * stride) * Cin + lh * 16;

#pragma unroll
  for (int kh = 0; kh < 3; ++kh) {
#pragma unroll
    for (int kw = 0; kw < 3; ++kw) {
      const __bf16* bbase = ibase + ((long)kh * Wip + kw) * Cin;
      const __bf16* abase = wbase + (long)(kh * 3 + kw) * nCic * 512;
      for (int c = 0; c < nCic; ++c) {
        v16bf A = *(const v16bf*)(abase + (long)c * 512);
        v16bf B = *(const v16bf*)(bbase + (long)c * 32);
        acc = __builtin_amdgcn_wmma_f32_16x16x32_bf16(
            false, A, false, B, (short)0, acc, false, false);
      }
    }
  }

  float mval = 1.0f;
  if (mask) mval = mask[((long)n * Ho + ho) * Wo + wo];
  const int Hop = Ho + 2, Wop = Wo + 2;
  const long ppos = ((long)n * Hop + (ho + 1)) * Wop + (wo + 1);   // padded NHWC position
  const long obase = ppos * Cout + cot * 16 + lh * 8;

  v8us rraw = (v8us)0;
  if (resid) rraw = *(const v8us*)((const unsigned short*)resid + obase);

  v8us obf;
  float ov[8];
#pragma unroll
  for (int j = 0; j < 8; ++j) {
    int co = cot * 16 + lh * 8 + j;
    float v = acc[j] * scale[co] + shift[co];
    v *= mval;
    if (resid) v += bf2f(rraw[j]);
    v = fmaxf(v, 0.0f);                     // every conv in this net ends in ReLU
    ov[j] = v;
    obf[j] = f2bf_bits(v);
  }
  if (out_bf) *(v8us*)((unsigned short*)out_bf + obase) = obf;
  if (out_f32) {
#pragma unroll
    for (int j = 0; j < 8; ++j) {
      int co = cot * 16 + lh * 8 + j;
      out_f32[(((long)n * Cout + co) * Ho + ho) * Wo + wo] = ov[j];
    }
  }
}

extern "C" void kernel_launch(void* const* d_in, const int* in_sizes, int n_in,
                              void* d_out, int out_size, void* d_ws, size_t ws_size,
                              hipStream_t stream) {
  (void)in_sizes; (void)n_in; (void)out_size; (void)ws_size;
  const float* x      = (const float*)d_in[0];
  const float* mask_p = (const float*)d_in[1];
  const float* w1 = (const float*)d_in[2];
  const float* s1 = (const float*)d_in[3];
  const float* t1 = (const float*)d_in[4];
  const float* wd2 = (const float*)d_in[5];
  const float* w2 = (const float*)d_in[6];
  const float* s2 = (const float*)d_in[7];
  const float* t2 = (const float*)d_in[8];
  const float* wd3 = (const float*)d_in[9];
  const float* w3 = (const float*)d_in[10];
  const float* s3 = (const float*)d_in[11];
  const float* t3 = (const float*)d_in[12];
  const float* wd4 = (const float*)d_in[13];
  const float* w4 = (const float*)d_in[14];
  const float* s4 = (const float*)d_in[15];
  const float* t4 = (const float*)d_in[16];
  const float* w5 = (const float*)d_in[17];
  const float* s5 = (const float*)d_in[18];
  const float* t5 = (const float*)d_in[19];

  float* out = (float*)d_out;
  char* ws = (char*)d_ws;

  // ---- workspace layout ----
  const long BUFB = 33817088;                    // 2*514*514*32 bf16 (padded stage-1 buffer)
  __bf16* bufX = (__bf16*)(ws);
  __bf16* bufA = (__bf16*)(ws + BUFB);
  __bf16* bufB = (__bf16*)(ws + 2 * BUFB);
  float* mask0 = (float*)(ws + 3 * BUFB);        // 2*512*512
  float* m2 = mask0 + 524288;                    // 2*256*256
  float* m3 = m2 + 131072;                       // 2*128*128
  float* m4 = m3 + 32768;                        // 2*64*64
  __bf16* wpk = (__bf16*)(ws + 3 * BUFB + (524288 + 131072 + 32768 + 8192) * 4);

  // ---- layer table ----
  const float* lw[23]; int lcin[23], lcout[23]; long woff[23];
  for (int i = 0; i < 5; ++i) { lw[i] = w1 + (long)i * 32 * 32 * 9; lcin[i] = 32; lcout[i] = 32; }
  lw[5] = wd2; lcin[5] = 32; lcout[5] = 64;
  for (int i = 0; i < 4; ++i) { lw[6 + i] = w2 + (long)i * 64 * 64 * 9; lcin[6 + i] = 64; lcout[6 + i] = 64; }
  lw[10] = wd3; lcin[10] = 64; lcout[10] = 128;
  for (int i = 0; i < 4; ++i) { lw[11 + i] = w3 + (long)i * 128 * 128 * 9; lcin[11 + i] = 128; lcout[11 + i] = 128; }
  lw[15] = wd4; lcin[15] = 128; lcout[15] = 256;
  for (int i = 0; i < 4; ++i) { lw[16 + i] = w4 + (long)i * 256 * 256 * 9; lcin[16 + i] = 256; lcout[16 + i] = 256; }
  for (int i = 0; i < 3; ++i) { lw[20 + i] = w5 + (long)i * 256 * 256 * 9; lcin[20 + i] = 256; lcout[20 + i] = 256; }
  {
    long a = 0;
    for (int i = 0; i < 23; ++i) { woff[i] = a; a += (long)lcout[i] * lcin[i] * 9; }
  }

  // ---- masks ----
  make_mask_kernel<<<(524288 + 255) / 256, 256, 0, stream>>>(mask_p, mask0, 524288);
  down_mask_kernel<<<(2 * 256 * 256 + 255) / 256, 256, 0, stream>>>(mask0, m2, 2, 512, 512);
  down_mask_kernel<<<(2 * 128 * 128 + 255) / 256, 256, 0, stream>>>(m2, m3, 2, 256, 256);
  down_mask_kernel<<<(2 * 64 * 64 + 255) / 256, 256, 0, stream>>>(m3, m4, 2, 128, 128);

  // ---- weight repack (A-fragment layout, bf16) ----
  for (int i = 0; i < 23; ++i) {
    long total = (long)lcout[i] * lcin[i] * 9;
    pack_weights_kernel<<<(int)((total + 255) / 256), 256, 0, stream>>>(
        lw[i], wpk + woff[i], lcin[i], lcout[i]);
  }

  // ---- masked input -> padded NHWC bf16 (halo zeroed first) ----
  {
    long ring = 2L * (2 * (512 + 2) + 2 * 512) * 32;
    ring_zero_kernel<<<(int)((ring + 255) / 256), 256, 0, stream>>>(bufX, 2, 512, 512, 32);
  }
  pack_input_kernel<<<(16777216 + 255) / 256, 256, 0, stream>>>(x, mask0, bufX, 2, 32, 512, 512);

  // ---- output slices ----
  float* x1o = out;
  float* x2o = x1o + 16777216;
  float* x3o = x2o + 8388608;
  float* x4o = x3o + 4194304;
  float* x5o = x4o + 2097152;

  auto conv = [&](const __bf16* in, int layer, const float* sc, const float* sh,
                  const float* msk, const __bf16* res, __bf16* obf, float* of32,
                  int Cin, int Cout, int Hi, int Wi, int stride) {
    int Ho = Hi / stride, Wo = Wi / stride;
    if (obf) {  // re-establish the zero halo of the (dead) output buffer
      long ring = 2L * (2 * (Wo + 2) + 2 * Ho) * Cout;
      ring_zero_kernel<<<(int)((ring + 255) / 256), 256, 0, stream>>>(obf, 2, Ho, Wo, Cout);
    }
    dim3 grid((Wo + 63) / 64, Ho, 2 * (Cout >> 4));
    conv3x3_wmma_kernel<<<grid, 128, 0, stream>>>(
        in, wpk + woff[layer], sc, sh, msk, res, obf, of32,
        Cin, Cout, Hi, Wi, Ho, Wo, stride);
  };

  // stage 1 @512^2, C32
  conv(bufX, 0, s1 + 0,   t1 + 0,   mask0, nullptr, bufA, nullptr, 32, 32, 512, 512, 1);
  conv(bufA, 1, s1 + 32,  t1 + 32,  mask0, nullptr, bufB, nullptr, 32, 32, 512, 512, 1);
  conv(bufB, 2, s1 + 64,  t1 + 64,  mask0, bufA,    bufX, nullptr, 32, 32, 512, 512, 1);
  conv(bufX, 3, s1 + 96,  t1 + 96,  mask0, nullptr, bufA, nullptr, 32, 32, 512, 512, 1);
  conv(bufA, 4, s1 + 128, t1 + 128, mask0, bufX,    bufB, x1o,     32, 32, 512, 512, 1);
  // stage 2 -> 256^2, C64
  conv(bufB, 5, s2 + 0,   t2 + 0,   m2, nullptr, bufX, nullptr, 32, 64, 512, 512, 2);
  conv(bufX, 6, s2 + 64,  t2 + 64,  m2, nullptr, bufA, nullptr, 64, 64, 256, 256, 1);
  conv(bufA, 7, s2 + 128, t2 + 128, m2, bufX,    bufB, nullptr, 64, 64, 256, 256, 1);
  conv(bufB, 8, s2 + 192, t2 + 192, m2, nullptr, bufX, nullptr, 64, 64, 256, 256, 1);
  conv(bufX, 9, s2 + 256, t2 + 256, m2, bufB,    bufA, x2o,     64, 64, 256, 256, 1);
  // stage 3 -> 128^2, C128
  conv(bufA, 10, s3 + 0,   t3 + 0,   m3, nullptr, bufX, nullptr, 64, 128, 256, 256, 2);
  conv(bufX, 11, s3 + 128, t3 + 128, m3, nullptr, bufB, nullptr, 128, 128, 128, 128, 1);
  conv(bufB, 12, s3 + 256, t3 + 256, m3, bufX,    bufA, nullptr, 128, 128, 128, 128, 1);
  conv(bufA, 13, s3 + 384, t3 + 384, m3, nullptr, bufX, nullptr, 128, 128, 128, 128, 1);
  conv(bufX, 14, s3 + 512, t3 + 512, m3, bufA,    bufB, x3o,     128, 128, 128, 128, 1);
  // stage 4 -> 64^2, C256
  conv(bufB, 15, s4 + 0,    t4 + 0,    m4, nullptr, bufX, nullptr, 128, 256, 128, 128, 2);
  conv(bufX, 16, s4 + 256,  t4 + 256,  m4, nullptr, bufA, nullptr, 256, 256, 64, 64, 1);
  conv(bufA, 17, s4 + 512,  t4 + 512,  m4, bufX,    bufB, nullptr, 256, 256, 64, 64, 1);
  conv(bufB, 18, s4 + 768,  t4 + 768,  m4, nullptr, bufX, nullptr, 256, 256, 64, 64, 1);
  conv(bufX, 19, s4 + 1024, t4 + 1024, m4, bufB,    bufA, x4o,     256, 256, 64, 64, 1);
  // stage 5 (dense) -> 32^2, C256
  conv(bufA, 20, s5 + 0,   t5 + 0,   nullptr, nullptr, bufX, nullptr, 256, 256, 64, 64, 2);
  conv(bufX, 21, s5 + 256, t5 + 256, nullptr, nullptr, bufB, nullptr, 256, 256, 32, 32, 1);
  conv(bufB, 22, s5 + 512, t5 + 512, nullptr, nullptr, nullptr, x5o,  256, 256, 32, 32, 1);
}
